// MoE_4372276707355
// MI455X (gfx1250) — matrix operand order repaired and checked by reading, not compiled, not run on hardware
//
#include <hip/hip_runtime.h>

#define TOK   8192   // B*S
#define DIM_I 1024
#define DIM_O 1024
#define NEXP  8
#define KT_I  32     // DIM_I/32  k-tiles
#define NT_O  64     // DIM_O/16  n-tiles
#define MT_T  512    // TOK/16    m-tiles

typedef __attribute__((ext_vector_type(16))) __bf16 v16bf;
typedef __attribute__((ext_vector_type(8)))  float  v8f;

static __device__ __forceinline__ __bf16 bf16_trunc(float f) {
  union { float f; unsigned u; } v; v.f = f;
  unsigned short h = (unsigned short)(v.u >> 16);
  __bf16 r;
  __builtin_memcpy(&r, &h, sizeof(r));
  return r;
}
static __device__ __forceinline__ float bf16_hi_part(float f) {
  union { float f; unsigned u; } v; v.f = f;
  v.u &= 0xffff0000u;
  return v.f;
}

// ---------------------------------------------------------------------------
// Kernel 1: per-token gate softmax.  One wave32 per token.
// ---------------------------------------------------------------------------
__global__ __launch_bounds__(256) void moe_gates(
    const float* __restrict__ x, const float* __restrict__ gw,
    float* __restrict__ g) {
  const int lane = threadIdx.x & 31;
  const int wid  = threadIdx.x >> 5;
  const int t    = blockIdx.x * 8 + wid;
  const int e    = lane & 7;
  const int c    = lane >> 3;

  const float* xr = x + (size_t)t * DIM_I;
  float p = 0.f;
  const int k0 = c * 256;
  for (int k = k0; k < k0 + 256; ++k)
    p += xr[k] * gw[k * NEXP + e];

  p += __shfl_xor(p, 8, 32);
  p += __shfl_xor(p, 16, 32);

  float m = p;
  m = fmaxf(m, __shfl_xor(m, 1, 32));
  m = fmaxf(m, __shfl_xor(m, 2, 32));
  m = fmaxf(m, __shfl_xor(m, 4, 32));
  float ex = __expf(p - m);
  float s = ex;
  s += __shfl_xor(s, 1, 32);
  s += __shfl_xor(s, 2, 32);
  s += __shfl_xor(s, 4, 32);
  if (lane < 8)
    g[(size_t)t * NEXP + lane] = ex / s;
}

// ---------------------------------------------------------------------------
// Kernel 2a: one-time x -> bf16 hi/lo planes in A-fragment-swizzled layout.
// Tile (mt,kt) = 16x32 f32.  One wave per tile; per lane: 16 elems -> 32B.
// Plane layout: tile id (mt*KT_I + kt), 1024B per tile, lane-major.
// ---------------------------------------------------------------------------
__global__ __launch_bounds__(256) void conv_x(
    const float* __restrict__ x, __bf16* __restrict__ xhi,
    __bf16* __restrict__ xlo) {
  const int lane = threadIdx.x & 31;
  const int wid  = threadIdx.x >> 5;
  const int tid  = blockIdx.x * 8 + wid;   // mt*KT_I + kt
  const int mt   = tid >> 5;
  const int kt   = tid & 31;
  const int row  = mt * 16 + (lane & 15);
  const int kb   = (lane >> 4) << 3;

  const float* ap = x + (size_t)row * DIM_I + kt * 32 + kb;
  float4 q0 = *reinterpret_cast<const float4*>(ap);
  float4 q1 = *reinterpret_cast<const float4*>(ap + 4);
  float4 q2 = *reinterpret_cast<const float4*>(ap + 16);
  float4 q3 = *reinterpret_cast<const float4*>(ap + 20);
  float fa[16] = {q0.x, q0.y, q0.z, q0.w, q1.x, q1.y, q1.z, q1.w,
                  q2.x, q2.y, q2.z, q2.w, q3.x, q3.y, q3.z, q3.w};
  v16bf hi, lo;
  #pragma unroll
  for (int j = 0; j < 16; ++j) {
    float h = bf16_hi_part(fa[j]);
    hi[j] = bf16_trunc(fa[j]);
    lo[j] = bf16_trunc(fa[j] - h);
  }
  const size_t off = (size_t)tid * 512 + lane * 16;
  *reinterpret_cast<v16bf*>(xhi + off) = hi;
  *reinterpret_cast<v16bf*>(xlo + off) = lo;
}

// ---------------------------------------------------------------------------
// Kernel 2b: one-time W -> bf16 hi/lo planes in B-fragment-swizzled layout.
// Tile (e,nt,kt) = 32(K)x16(N) f32.  Plane layout: ((e*NT_O+nt)*KT_I+kt),
// 1024B per tile (so the main k-loop walks contiguous tiles).
// ---------------------------------------------------------------------------
__global__ __launch_bounds__(256) void conv_w(
    const float* __restrict__ w, __bf16* __restrict__ whi,
    __bf16* __restrict__ wlo) {
  const int lane = threadIdx.x & 31;
  const int wid  = threadIdx.x >> 5;
  const int tid  = blockIdx.x * 8 + wid;   // (e*NT_O + nt)*KT_I + kt
  const int e    = tid >> 11;
  const int nt   = (tid >> 5) & 63;
  const int kt   = tid & 31;
  const int n    = nt * 16 + (lane & 15);
  const int kb   = (lane >> 4) << 3;

  const float* bp =
      w + (size_t)e * DIM_I * DIM_O + (size_t)(kt * 32 + kb) * DIM_O + n;
  v16bf hi, lo;
  #pragma unroll
  for (int j = 0; j < 16; ++j) {
    float f = bp[(size_t)(j + (j & 8)) * DIM_O];
    float h = bf16_hi_part(f);
    hi[j] = bf16_trunc(f);
    lo[j] = bf16_trunc(f - h);
  }
  const size_t off = (size_t)tid * 512 + lane * 16;
  *reinterpret_cast<v16bf*>(whi + off) = hi;
  *reinterpret_cast<v16bf*>(wlo + off) = lo;
}

// ---------------------------------------------------------------------------
// Kernel 3: main all-expert GEMM + gate-weighted combine on pre-swizzled
// bf16 hi/lo planes.  Hot loop = b128 loads + 12 WMMAs, no conversions.
// Block: 8 waves, tile 128(M) x 64(N); wave tile 32x32 (2x2 fragments).
// ---------------------------------------------------------------------------
__global__ __launch_bounds__(256, 1) void moe_expert_gemm(
    const __bf16* __restrict__ xhi, const __bf16* __restrict__ xlo,
    const __bf16* __restrict__ whi, const __bf16* __restrict__ wlo,
    const float* __restrict__ bias, const float* __restrict__ g,
    float* __restrict__ out) {
  const int lane  = threadIdx.x & 31;
  const int wid   = threadIdx.x >> 5;
  const int mBase = blockIdx.x * 128 + (wid >> 1) * 32;
  const int nBase = blockIdx.y * 64  + (wid & 1) * 32;
  const int mt0   = mBase >> 4;     // first global m-tile
  const int nt0   = nBase >> 4;     // first global n-tile
  const int laneOff = lane * 16;
  const int colB  = lane & 15;
  const int mHi   = (lane >> 4) << 3;

  const __bf16* xa_hi = xhi + (size_t)mt0 * KT_I * 512 + laneOff;
  const __bf16* xa_lo = xlo + (size_t)mt0 * KT_I * 512 + laneOff;

  v8f fin[2][2];
  #pragma unroll
  for (int mt = 0; mt < 2; ++mt)
    #pragma unroll
    for (int nt = 0; nt < 2; ++nt)
      fin[mt][nt] = {};

  for (int e = 0; e < NEXP; ++e) {
    const size_t wtile = (size_t)(e * NT_O + nt0) * KT_I * 512;
    const __bf16* wb_hi = whi + wtile + laneOff;
    const __bf16* wb_lo = wlo + wtile + laneOff;

    v8f acc[2][2];
    #pragma unroll
    for (int mt = 0; mt < 2; ++mt)
      #pragma unroll
      for (int nt = 0; nt < 2; ++nt)
        acc[mt][nt] = {};

    for (int kt = 0; kt < KT_I; ++kt) {
      v16bf ahi[2], alo[2], bhi[2], blo[2];
      ahi[0] = *reinterpret_cast<const v16bf*>(xa_hi + (size_t)kt * 512);
      ahi[1] = *reinterpret_cast<const v16bf*>(xa_hi + (size_t)(KT_I + kt) * 512);
      alo[0] = *reinterpret_cast<const v16bf*>(xa_lo + (size_t)kt * 512);
      alo[1] = *reinterpret_cast<const v16bf*>(xa_lo + (size_t)(KT_I + kt) * 512);
      bhi[0] = *reinterpret_cast<const v16bf*>(wb_hi + (size_t)kt * 512);
      bhi[1] = *reinterpret_cast<const v16bf*>(wb_hi + (size_t)(KT_I + kt) * 512);
      blo[0] = *reinterpret_cast<const v16bf*>(wb_lo + (size_t)kt * 512);
      blo[1] = *reinterpret_cast<const v16bf*>(wb_lo + (size_t)(KT_I + kt) * 512);

      #pragma unroll
      for (int mt = 0; mt < 2; ++mt)
        #pragma unroll
        for (int nt = 0; nt < 2; ++nt) {
          acc[mt][nt] = __builtin_amdgcn_wmma_f32_16x16x32_bf16(
              false, ahi[mt], false, bhi[nt], (short)0, acc[mt][nt], false, false);
          acc[mt][nt] = __builtin_amdgcn_wmma_f32_16x16x32_bf16(
              false, ahi[mt], false, blo[nt], (short)0, acc[mt][nt], false, false);
          acc[mt][nt] = __builtin_amdgcn_wmma_f32_16x16x32_bf16(
              false, alo[mt], false, bhi[nt], (short)0, acc[mt][nt], false, false);
        }
    }

    #pragma unroll
    for (int mt = 0; mt < 2; ++mt) {
      float gv[8];
      #pragma unroll
      for (int r = 0; r < 8; ++r)
        gv[r] = g[(size_t)(mBase + mt * 16 + mHi + r) * NEXP + e];
      #pragma unroll
      for (int nt = 0; nt < 2; ++nt) {
        float bv = bias[e * DIM_O + nBase + nt * 16 + colB];
        #pragma unroll
        for (int r = 0; r < 8; ++r)
          fin[mt][nt][r] += gv[r] * (acc[mt][nt][r] + bv);
      }
    }
  }

  #pragma unroll
  for (int mt = 0; mt < 2; ++mt)
    #pragma unroll
    for (int nt = 0; nt < 2; ++nt)
      #pragma unroll
      for (int r = 0; r < 8; ++r)
        out[(size_t)(mBase + mt * 16 + mHi + r) * DIM_O +
            nBase + nt * 16 + colB] = fin[mt][nt][r];
}

// ---------------------------------------------------------------------------
// Fallback (small ws): fused-conversion GEMM (round-1 style).
// ---------------------------------------------------------------------------
__global__ __launch_bounds__(256, 1) void moe_expert_gemm_direct(
    const float* __restrict__ x, const float* __restrict__ w,
    const float* __restrict__ bias, const float* __restrict__ g,
    float* __restrict__ out) {
  const int lane  = threadIdx.x & 31;
  const int wid   = threadIdx.x >> 5;
  const int mBase = blockIdx.x * 128 + (wid >> 1) * 32;
  const int nBase = blockIdx.y * 64  + (wid & 1) * 32;
  const int rowA = lane & 15;
  const int kbA  = (lane >> 4) << 3;
  const int kbB  = (lane >> 4) << 3;
  const int colB = lane & 15;
  const int mHi  = (lane >> 4) << 3;

  v8f fin[2][2];
  #pragma unroll
  for (int mt = 0; mt < 2; ++mt)
    #pragma unroll
    for (int nt = 0; nt < 2; ++nt)
      fin[mt][nt] = {};

  for (int e = 0; e < NEXP; ++e) {
    const float* we = w + (size_t)e * DIM_I * DIM_O;
    v8f acc[2][2];
    #pragma unroll
    for (int mt = 0; mt < 2; ++mt)
      #pragma unroll
      for (int nt = 0; nt < 2; ++nt)
        acc[mt][nt] = {};

    for (int k0 = 0; k0 < DIM_I; k0 += 32) {
      v16bf ahi[2], alo[2], bhi[2], blo[2];
      #pragma unroll
      for (int mt = 0; mt < 2; ++mt) {
        const float* ap =
            x + (size_t)(mBase + mt * 16 + rowA) * DIM_I + k0 + kbA;
        float4 q0 = *reinterpret_cast<const float4*>(ap);
        float4 q1 = *reinterpret_cast<const float4*>(ap + 4);
        float4 q2 = *reinterpret_cast<const float4*>(ap + 16);
        float4 q3 = *reinterpret_cast<const float4*>(ap + 20);
        float fa[16] = {q0.x, q0.y, q0.z, q0.w, q1.x, q1.y, q1.z, q1.w,
                        q2.x, q2.y, q2.z, q2.w, q3.x, q3.y, q3.z, q3.w};
        #pragma unroll
        for (int j = 0; j < 16; ++j) {
          float h = bf16_hi_part(fa[j]);
          ahi[mt][j] = bf16_trunc(fa[j]);
          alo[mt][j] = bf16_trunc(fa[j] - h);
        }
      }
      #pragma unroll
      for (int nt = 0; nt < 2; ++nt) {
        const float* bp =
            we + (size_t)(k0 + kbB) * DIM_O + nBase + nt * 16 + colB;
        #pragma unroll
        for (int j = 0; j < 16; ++j) {
          float f = bp[(size_t)(j + (j & 8)) * DIM_O];
          float h = bf16_hi_part(f);
          bhi[nt][j] = bf16_trunc(f);
          blo[nt][j] = bf16_trunc(f - h);
        }
      }
      #pragma unroll
      for (int mt = 0; mt < 2; ++mt)
        #pragma unroll
        for (int nt = 0; nt < 2; ++nt) {
          acc[mt][nt] = __builtin_amdgcn_wmma_f32_16x16x32_bf16(
              false, ahi[mt], false, bhi[nt], (short)0, acc[mt][nt], false, false);
          acc[mt][nt] = __builtin_amdgcn_wmma_f32_16x16x32_bf16(
              false, ahi[mt], false, blo[nt], (short)0, acc[mt][nt], false, false);
          acc[mt][nt] = __builtin_amdgcn_wmma_f32_16x16x32_bf16(
              false, alo[mt], false, bhi[nt], (short)0, acc[mt][nt], false, false);
        }
    }

    #pragma unroll
    for (int mt = 0; mt < 2; ++mt) {
      float gv[8];
      #pragma unroll
      for (int r = 0; r < 8; ++r)
        gv[r] = g[(size_t)(mBase + mt * 16 + mHi + r) * NEXP + e];
      #pragma unroll
      for (int nt = 0; nt < 2; ++nt) {
        float bv = bias[e * DIM_O + nBase + nt * 16 + colB];
        #pragma unroll
        for (int r = 0; r < 8; ++r)
          fin[mt][nt][r] += gv[r] * (acc[mt][nt][r] + bv);
      }
    }
  }

  #pragma unroll
  for (int mt = 0; mt < 2; ++mt)
    #pragma unroll
    for (int nt = 0; nt < 2; ++nt)
      #pragma unroll
      for (int r = 0; r < 8; ++r)
        out[(size_t)(mBase + mt * 16 + mHi + r) * DIM_O +
            nBase + nt * 16 + colB] = fin[mt][nt][r];
}

// ---------------------------------------------------------------------------
extern "C" void kernel_launch(void* const* d_in, const int* in_sizes, int n_in,
                              void* d_out, int out_size, void* d_ws, size_t ws_size,
                              hipStream_t stream) {
  const float* x  = (const float*)d_in[0];  // [4,2048,1024]
  const float* ew = (const float*)d_in[1];  // [8,1024,1024]
  const float* eb = (const float*)d_in[2];  // [8,1024]
  const float* gw = (const float*)d_in[3];  // [1024,8]
  float* out   = (float*)d_out;
  float* gates = (float*)d_ws;

  const size_t GATES_BYTES = (size_t)TOK * NEXP * sizeof(float);   // 256 KB
  const size_t PLANE = (size_t)TOK * DIM_I * sizeof(__bf16);       // 16 MB (==E*I*O bf16)
  const size_t need = GATES_BYTES + 4 * PLANE;

  moe_gates<<<dim3(TOK / 8), dim3(256), 0, stream>>>(x, gw, gates);

  if (ws_size >= need) {
    char* base = (char*)d_ws + GATES_BYTES;
    __bf16* xhi = (__bf16*)(base);
    __bf16* xlo = (__bf16*)(base + PLANE);
    __bf16* whi = (__bf16*)(base + 2 * PLANE);
    __bf16* wlo = (__bf16*)(base + 3 * PLANE);
    conv_x<<<dim3(MT_T * KT_I / 8), dim3(256), 0, stream>>>(x, xhi, xlo);
    conv_w<<<dim3(NEXP * NT_O * KT_I / 8), dim3(256), 0, stream>>>(ew, whi, wlo);
    moe_expert_gemm<<<dim3(TOK / 128, DIM_O / 64), dim3(256), 0, stream>>>(
        xhi, xlo, whi, wlo, eb, gates, out);
  } else {
    moe_expert_gemm_direct<<<dim3(TOK / 128, DIM_O / 64), dim3(256), 0, stream>>>(
        x, ew, eb, gates, out);
  }
}
